// RNN_52982716563926
// MI455X (gfx1250) — matrix operand order repaired and checked by reading, not compiled, not run on hardware
//
#include <hip/hip_runtime.h>

#define SEQ   512
#define BATCH 128
#define DIN   512
#define DH    1024
#define DOUT  256
#define KCAT  (DIN + DH)   // 1536
#define NWG   16           // recurrence blocks (one per WGP; 192KB LDS each)
#define XC    (DIN / 32)   // 16 x-chunks
#define HC    (DH / 32)    // 32 h-chunks

typedef __attribute__((ext_vector_type(16))) __bf16 v16bf;
typedef __attribute__((ext_vector_type(8)))  __bf16 v8bf;
typedef __attribute__((ext_vector_type(8)))  float  v8f;
typedef __attribute__((ext_vector_type(4)))  float  v4f;

// ---------------- workspace layout (bytes) ----------------
static constexpr size_t OFF_H0     = 0;                                   // bf16 [BATCH][DH]
static constexpr size_t OFF_H1     = OFF_H0 + (size_t)BATCH * DH * 2;     // bf16 [BATCH][DH]
static constexpr size_t OFF_LOGITS = OFF_H1 + (size_t)BATCH * DH * 2;     // f32  [BATCH][DOUT]
static constexpr size_t OFF_BAR    = OFF_LOGITS + (size_t)BATCH * DOUT * 4; // 2 x u32

// ---------------- helpers ----------------
__device__ __forceinline__ v16bf cat8(v8bf lo, v8bf hi) {
    return __builtin_shufflevector(lo, hi, 0,1,2,3,4,5,6,7,8,9,10,11,12,13,14,15);
}

__device__ __forceinline__ v16bf cvt16(v4f f0, v4f f1, v4f f2, v4f f3) {
    v16bf v;
    v[0]=(__bf16)f0[0];  v[1]=(__bf16)f0[1];  v[2]=(__bf16)f0[2];  v[3]=(__bf16)f0[3];
    v[4]=(__bf16)f1[0];  v[5]=(__bf16)f1[1];  v[6]=(__bf16)f1[2];  v[7]=(__bf16)f1[3];
    v[8]=(__bf16)f2[0];  v[9]=(__bf16)f2[1];  v[10]=(__bf16)f2[2]; v[11]=(__bf16)f2[3];
    v[12]=(__bf16)f3[0]; v[13]=(__bf16)f3[1]; v[14]=(__bf16)f3[2]; v[15]=(__bf16)f3[3];
    return v;
}

__device__ __forceinline__ float fast_tanh(float v) {
#if __has_builtin(__builtin_amdgcn_tanhf)
    return __builtin_amdgcn_tanhf(v);      // gfx1250 v_tanh_f32 (TRANS unit)
#else
    return tanhf(v);
#endif
}

// A-fragment from fp32 row (16 lanes = rows, lane half selects K-subgroup)
__device__ __forceinline__ v16bf loadA_f32(const float* rowp, int koff, int half) {
    const float* p0 = rowp + koff + half * 8;
    v4f f0 = *(const v4f*)(p0);
    v4f f1 = *(const v4f*)(p0 + 4);
    v4f f2 = *(const v4f*)(p0 + 16);
    v4f f3 = *(const v4f*)(p0 + 20);
    return cvt16(f0, f1, f2, f3);
}

// A-fragment from bf16 row
__device__ __forceinline__ v16bf loadA_bf16(const __bf16* rowp, int koff, int half) {
    v8bf lo = *(const v8bf*)(rowp + koff + half * 8);
    v8bf hi = *(const v8bf*)(rowp + koff + 16 + half * 8);
    return cat8(lo, hi);
}

// Load all 4 B-fragments of K-chunk `c` from LDS, then issue 4 WMMAs.
__device__ __forceinline__ void wmma4(const __bf16* lds, int c, int lane,
                                      v16bf a, v8f acc[4]) {
    v16bf b0 = *(const v16bf*)&lds[(size_t)(0 * 48 + c) * 512 + lane * 16];
    v16bf b1 = *(const v16bf*)&lds[(size_t)(1 * 48 + c) * 512 + lane * 16];
    v16bf b2 = *(const v16bf*)&lds[(size_t)(2 * 48 + c) * 512 + lane * 16];
    v16bf b3 = *(const v16bf*)&lds[(size_t)(3 * 48 + c) * 512 + lane * 16];
    acc[0] = __builtin_amdgcn_wmma_f32_16x16x32_bf16(false, a, false, b0, (short)0, acc[0], false, false);
    acc[1] = __builtin_amdgcn_wmma_f32_16x16x32_bf16(false, a, false, b1, (short)0, acc[1], false, false);
    acc[2] = __builtin_amdgcn_wmma_f32_16x16x32_bf16(false, a, false, b2, (short)0, acc[2], false, false);
    acc[3] = __builtin_amdgcn_wmma_f32_16x16x32_bf16(false, a, false, b3, (short)0, acc[3], false, false);
}

__device__ __forceinline__ void grid_barrier(unsigned* bar, unsigned* gen, unsigned nwg) {
    __syncthreads();
    if (threadIdx.x == 0) {
        unsigned g = __hip_atomic_load(gen, __ATOMIC_RELAXED, __HIP_MEMORY_SCOPE_AGENT);
        unsigned a = __hip_atomic_fetch_add(bar, 1u, __ATOMIC_ACQ_REL, __HIP_MEMORY_SCOPE_AGENT);
        if (a == nwg - 1u) {
            __hip_atomic_store(bar, 0u, __ATOMIC_RELAXED, __HIP_MEMORY_SCOPE_AGENT);
            __hip_atomic_fetch_add(gen, 1u, __ATOMIC_RELEASE, __HIP_MEMORY_SCOPE_AGENT);
        } else {
            while (__hip_atomic_load(gen, __ATOMIC_ACQUIRE, __HIP_MEMORY_SCOPE_AGENT) == g) {
                __builtin_amdgcn_s_sleep(1);
            }
        }
    }
    __syncthreads();
}

// ---------------- kernel 0: hidden fp32 -> bf16 ----------------
__global__ void cvt_hidden(const float* __restrict__ hidden, __bf16* __restrict__ h0) {
    const size_t n = (size_t)BATCH * DH;
    const size_t stride = (size_t)gridDim.x * blockDim.x;
    for (size_t i = (size_t)blockIdx.x * blockDim.x + threadIdx.x; i < n; i += stride)
        h0[i] = (__bf16)hidden[i];
}

// ---------------- kernel 1: persistent recurrence ----------------
__global__ __launch_bounds__(256) void rnn_recurrence(
    const float* __restrict__ x,       // [SEQ][BATCH][DIN]
    const float* __restrict__ Wi2h,    // [DH][KCAT] fp32 (row-major == B^T layout)
    const float* __restrict__ b_i2h,   // [DH]
    __bf16* __restrict__ h0,           // ping
    __bf16* __restrict__ h1,           // pong
    float* __restrict__ hfinal_out,    // d_out + BATCH*DOUT
    unsigned* __restrict__ bar)
{
    extern __shared__ __bf16 lds[];              // 4 * 48 * 512 bf16 = 192 KB
    const int lane  = threadIdx.x & 31;
    const int wave  = threadIdx.x >> 5;          // 0..7 == M-tile
    const int half  = lane >> 4;
    const int l16   = lane & 15;
    const int nBaseT = blockIdx.x * 4;           // first global N-tile of this block
    const int rowA  = wave * 16 + l16;           // A-fragment row this lane feeds

    // ---- stage weight fragments into LDS, fp32 -> bf16, fragment-swizzled ----
    for (int it = wave; it < 4 * 48; it += 8) {
        const int nLocal = it / 48;
        const int c      = it % 48;
        const float* src = Wi2h + (size_t)((nBaseT + nLocal) * 16 + l16) * KCAT
                                + c * 32 + half * 8;
        v4f f0 = *(const v4f*)(src);
        v4f f1 = *(const v4f*)(src + 4);
        v4f f2 = *(const v4f*)(src + 16);
        v4f f3 = *(const v4f*)(src + 20);
        *(v16bf*)&lds[(size_t)it * 512 + lane * 16] = cvt16(f0, f1, f2, f3);
    }
    __syncthreads();

    // ---- loop-invariant bias values for this wave's 4 column groups ----
    float bias[4];
    #pragma unroll
    for (int n = 0; n < 4; ++n) bias[n] = b_i2h[(nBaseT + n) * 16 + l16];

    for (int t = 0; t < SEQ; ++t) {
        const __bf16* hprev = (t & 1) ? h1 : h0;
        __bf16*       hnext = (t & 1) ? h0 : h1;
        const float*  xt    = x + ((size_t)t * BATCH + rowA) * DIN;
        const __bf16* hrow  = hprev + (size_t)rowA * DH;

        if (t + 1 < SEQ) __builtin_prefetch(xt + (size_t)BATCH * DIN, 0, 0);

        v8f acc[4];
        #pragma unroll
        for (int n = 0; n < 4; ++n) acc[n] = (v8f){0.f,0.f,0.f,0.f,0.f,0.f,0.f,0.f};

        // ---- fully unrolled, software-pipelined K loop (48 chunks, 192 WMMAs).
        // Full unroll folds the A-source select at compile time: straight-line
        // body, fixed accumulator registers, LDS addresses become immediates.
        v16bf a_cur = loadA_f32(xt, 0, half);
        #pragma unroll
        for (int c = 0; c < XC + HC; ++c) {
            v16bf a_nxt;
            if (c + 1 < XC)            a_nxt = loadA_f32(xt, (c + 1) * 32, half);
            else if (c + 1 < XC + HC)  a_nxt = loadA_bf16(hrow, (c + 1 - XC) * 32, half);
            else                       a_nxt = a_cur;
            wmma4(lds, c, lane, a_cur, acc);
            a_cur = a_nxt;
        }

        // ---- epilogue: bias + tanh; bf16 h_next, fp32 h_final on last step ----
        #pragma unroll
        for (int n = 0; n < 4; ++n) {
            const int col = (nBaseT + n) * 16 + l16;
            #pragma unroll
            for (int r = 0; r < 8; ++r) {
                const int row = wave * 16 + r + half * 8;
                const float v = fast_tanh(acc[n][r] + bias[n]);
                hnext[(size_t)row * DH + col] = (__bf16)v;
                if (t == SEQ - 1) hfinal_out[(size_t)row * DH + col] = v;
            }
        }
        grid_barrier(bar, bar + 1, gridDim.x);
    }
}

// ---------------- kernel 2: h_final @ W_h2o^T + b (128 waves, one 16x16 tile each) ----------------
__global__ __launch_bounds__(256) void rnn_h2o(
    const __bf16* __restrict__ h,     // [BATCH][DH] bf16 (hbuf0 after 512 steps)
    const float* __restrict__ Wh2o,   // [DOUT][DH] fp32 (row-major == B^T layout)
    const float* __restrict__ b_h2o,
    float* __restrict__ logits)       // [BATCH][DOUT]
{
    const int lane  = threadIdx.x & 31;
    const int w     = blockIdx.x * 8 + (threadIdx.x >> 5); // 0..127
    const int mTile = w >> 4;   // 0..7
    const int nTile = w & 15;   // 0..15
    const int half  = lane >> 4;
    const int l16   = lane & 15;
    const __bf16* arow = h    + (size_t)(mTile * 16 + l16) * DH;
    const float*  brow = Wh2o + (size_t)(nTile * 16 + l16) * DH;

    v8f acc = (v8f){0.f,0.f,0.f,0.f,0.f,0.f,0.f,0.f};
    for (int c = 0; c < DH / 32; ++c) {
        const int koff = c * 32;
        v16bf a = loadA_bf16(arow, koff, half);
        const float* p = brow + koff + half * 8;
        v16bf b = cvt16(*(const v4f*)(p), *(const v4f*)(p + 4),
                        *(const v4f*)(p + 16), *(const v4f*)(p + 20));
        acc = __builtin_amdgcn_wmma_f32_16x16x32_bf16(false, a, false, b, (short)0, acc, false, false);
    }
    const int col  = nTile * 16 + l16;
    const float bv = b_h2o[col];
    #pragma unroll
    for (int r = 0; r < 8; ++r) {
        const int row = mTile * 16 + r + half * 8;
        logits[(size_t)row * DOUT + col] = acc[r] + bv;
    }
}

// ---------------- kernel 3: row softmax (block per row) ----------------
__global__ __launch_bounds__(256) void softmax_rows(const float* __restrict__ logits,
                                                    float* __restrict__ out) {
    __shared__ float red[DOUT];
    const int r = blockIdx.x;
    const int c = threadIdx.x;
    const float v = logits[(size_t)r * DOUT + c];
    red[c] = v; __syncthreads();
    for (int s = DOUT / 2; s > 0; s >>= 1) {
        if (c < s) red[c] = fmaxf(red[c], red[c + s]);
        __syncthreads();
    }
    const float m = red[0]; __syncthreads();
    const float e = __expf(v - m);
    red[c] = e; __syncthreads();
    for (int s = DOUT / 2; s > 0; s >>= 1) {
        if (c < s) red[c] += red[c + s];
        __syncthreads();
    }
    out[(size_t)r * DOUT + c] = e / red[0];
}

// ---------------- host launcher ----------------
extern "C" void kernel_launch(void* const* d_in, const int* in_sizes, int n_in,
                              void* d_out, int out_size, void* d_ws, size_t ws_size,
                              hipStream_t stream) {
    const float* x      = (const float*)d_in[0];
    const float* hidden = (const float*)d_in[1];
    const float* Wi2h   = (const float*)d_in[2];
    const float* bi2h   = (const float*)d_in[3];
    const float* Wh2o   = (const float*)d_in[4];
    const float* bh2o   = (const float*)d_in[5];
    float* out = (float*)d_out;                       // [32768 softmax][131072 h_final]
    char* ws = (char*)d_ws;

    __bf16*   h0     = (__bf16*)(ws + OFF_H0);
    __bf16*   h1     = (__bf16*)(ws + OFF_H1);
    float*    logits = (float*)(ws + OFF_LOGITS);
    unsigned* bar    = (unsigned*)(ws + OFF_BAR);

    hipMemsetAsync(bar, 0, 2 * sizeof(unsigned), stream);   // graph-capturable
    cvt_hidden<<<64, 256, 0, stream>>>(hidden, h0);

    const size_t ldsBytes = (size_t)4 * 48 * 512 * sizeof(__bf16);  // 192 KB
    rnn_recurrence<<<NWG, 256, ldsBytes, stream>>>(x, Wi2h, bi2h, h0, h1,
                                                   out + (size_t)BATCH * DOUT, bar);
    // 512 steps is even -> final h lands in h0
    rnn_h2o<<<16, 256, 0, stream>>>(h0, Wh2o, bh2o, logits);
    softmax_rows<<<BATCH, DOUT, 0, stream>>>(logits, out);
}
